// ExtractorMLP_15616501088596
// MI455X (gfx1250) — matrix operand order repaired and checked by reading, not compiled, not run on hardware
//
#include <hip/hip_runtime.h>

// ============================================================================
// ExtractorMLP on MI455X (gfx1250, wave32, WMMA).
//
// Pipeline (all recompute, no 1GB intermediates):
//   P0: cast emb->bf16, build W1^T / W2^T in bf16, zero counters
//   P1: count edges per graph          P2: exclusive scan (1 block)
//   P3: scatter edges into graph-sorted order (perm/seg/col/row)
//   S1: 1 workgroup per graph: GEMM1 via v_wmma_f32_16x16x32_bf16,
//       register-accumulated sum/sumsq -> mean1/rstd1  (no global atomics)
//   S2: 1 workgroup per graph: recompute GEMM1, norm+relu, GEMM2 (WMMA),
//       register stats -> mean2/rstd2
//   F : tile sorted edges: recompute GEMM1+norm+relu+GEMM2+norm+relu,
//       dot with W3, scatter to original edge order via perm.
// ============================================================================

#define E_EDGES 1000000
#define NNODES  50000
#define NGRAPH  512
#define EPSI    1e-5f

typedef __bf16 bf16;
typedef __attribute__((ext_vector_type(16))) __bf16 v16bf;
typedef __attribute__((ext_vector_type(8)))  __bf16 v8bf;
typedef __attribute__((ext_vector_type(8)))  float  v8f;

union U16 { v16bf v; v8bf h[2]; };

// ---------------------------------------------------------------------------
// One 16x16 C tile: A(16 x kTot, LDS row-major) * B(kTot x 16) where B is
// stored transposed (bT[n][k], row stride kTot).  kTot in {128, 256}.
// A fragment: lanes 0-15 hold K 0..15, lanes 16-31 hold K 16..31 (contiguous,
// matching B).  B: 16 contiguous K per lane-half.
// ---------------------------------------------------------------------------
__device__ __forceinline__ v8f wmma_tile(const bf16* aT, const bf16* bT,
                                         int kTot, int n0, int lane) {
  v8f c = {0.f, 0.f, 0.f, 0.f, 0.f, 0.f, 0.f, 0.f};
  const int m  = lane & 15;
  const int hi = lane >> 4;
  const bf16* aBase = aT + m * kTot + hi * 16;
  const bf16* bBase = bT + (n0 + m) * kTot + hi * 16;
  const int chunks = kTot >> 5;
  for (int kc = 0; kc < chunks; ++kc) {
    U16 a;
    a.h[0] = *(const v8bf*)(aBase + kc * 32);
    a.h[1] = *(const v8bf*)(aBase + kc * 32 + 8);
    v16bf b = *(const v16bf*)(bBase + kc * 32);
    c = __builtin_amdgcn_wmma_f32_16x16x32_bf16(false, a.v, false, b,
                                                (short)0, c, false, false);
  }
  return c;
}

// Gather a 16-edge x 128-feature bf16 A tile ( emb[col] || emb[row] ) into LDS.
// 256 threads, 16 bytes each. Rows beyond [base,end) are zero-filled.
__device__ __forceinline__ void stage_atile(bf16* aT, const bf16* embB,
                                            const int* sCol, const int* sRow,
                                            int base, int end, int t) {
  const int mrow = t >> 4, part = t & 15;
  const int pos = base + mrow;
  uint4 val = make_uint4(0u, 0u, 0u, 0u);
  if (pos < end) {
    const int node = (part < 8) ? sCol[pos] : sRow[pos];
    val = *(const uint4*)(embB + (size_t)node * 64 + (part & 7) * 8);
  }
  *(uint4*)(aT + mrow * 128 + part * 8) = val;
}

// ---------------------------------------------------------------------------
// P0: dtype prep + counter clear
// ---------------------------------------------------------------------------
__global__ void k_prep(const float* __restrict__ emb, const float* __restrict__ W1,
                       const float* __restrict__ W2, bf16* embB, bf16* w1t,
                       bf16* w2t, int* counts) {
  const long i = (long)blockIdx.x * blockDim.x + threadIdx.x;
  const long stride = (long)gridDim.x * blockDim.x;
  for (long x = i; x < (long)NNODES * 64; x += stride) embB[x] = (bf16)emb[x];
  for (long x = i; x < 256 * 128; x += stride) {
    const int n = (int)(x >> 7), k = (int)(x & 127);
    w1t[x] = (bf16)W1[k * 256 + n];                 // W1 is [128][256]
  }
  for (long x = i; x < 64 * 256; x += stride) {
    const int n = (int)(x >> 8), k = (int)(x & 255);
    w2t[x] = (bf16)W2[k * 64 + n];                  // W2 is [256][64]
  }
  for (long x = i; x < NGRAPH; x += stride) counts[x] = 0;
}

// P1: histogram of edges per graph
__global__ void k_count(const int* __restrict__ ei, const int* __restrict__ batch,
                        int* counts) {
  const int i = blockIdx.x * blockDim.x + threadIdx.x;
  const int stride = gridDim.x * blockDim.x;
  for (int e = i; e < E_EDGES; e += stride)
    atomicAdd(&counts[batch[ei[e]]], 1);
}

// P2: exclusive scan over 512 counters (single 512-thread block)
__global__ __launch_bounds__(512) void k_scan(const int* __restrict__ counts,
                                              int* offs, int* cursor) {
  __shared__ int s[NGRAPH];
  const int t = threadIdx.x;
  s[t] = counts[t];
  __syncthreads();
  for (int d = 1; d < NGRAPH; d <<= 1) {
    const int v = (t >= d) ? s[t - d] : 0;
    __syncthreads();
    s[t] += v;
    __syncthreads();
  }
  offs[t + 1] = s[t];
  if (t == 0) offs[0] = 0;
  cursor[t] = s[t] - counts[t];
}

// P3: scatter edges into graph-sorted order
__global__ void k_scatter(const int* __restrict__ ei, const int* __restrict__ batch,
                          int* cursor, int* perm, int* segS, int* sCol, int* sRow) {
  const int i = blockIdx.x * blockDim.x + threadIdx.x;
  const int stride = gridDim.x * blockDim.x;
  for (int e = i; e < E_EDGES; e += stride) {
    const int c = ei[e], r = ei[E_EDGES + e];
    const int g = batch[c];
    const int pos = atomicAdd(&cursor[g], 1);
    perm[pos] = e; segS[pos] = g; sCol[pos] = c; sRow[pos] = r;
  }
}

// ---------------------------------------------------------------------------
// S1: per-graph stats of x1 = [emb[c]||emb[r]] @ W1 + b1   (256 channels)
// One workgroup per graph; stats accumulate in registers (lane channel fixed).
// ---------------------------------------------------------------------------
__global__ __launch_bounds__(256) void k_stats1(
    const bf16* __restrict__ embB, const bf16* __restrict__ w1tG,
    const float* __restrict__ b1, const int* __restrict__ sCol,
    const int* __restrict__ sRow, const int* __restrict__ offs,
    float* mean1, float* rstd1) {
  __shared__ __align__(32) bf16 sW1T[256 * 128];
  __shared__ __align__(32) bf16 sA[16 * 128];
  __shared__ float sStat[512];
  const int t = threadIdx.x, g = blockIdx.x;
  for (int i = t; i < 256 * 128 / 8; i += 256)
    ((uint4*)sW1T)[i] = ((const uint4*)w1tG)[i];
  sStat[t] = 0.f; sStat[t + 256] = 0.f;
  __syncthreads();

  const int base = offs[g], end = offs[g + 1];
  const int wave = t >> 5, lane = t & 31;
  const int nA = wave * 32 + (lane & 15), nB = nA + 16;
  const float b1A = b1[nA], b1B = b1[nB];
  const int mbase = (lane < 16) ? 0 : 8;
  float sumA = 0.f, sqA = 0.f, sumB = 0.f, sqB = 0.f;

  for (int tb = base; tb < end; tb += 16) {
    const int tcnt = min(16, end - tb);
    stage_atile(sA, embB, sCol, sRow, tb, end, t);
    __syncthreads();
    v8f cA = wmma_tile(sA, sW1T, 128, wave * 32, lane);
    v8f cB = wmma_tile(sA, sW1T, 128, wave * 32 + 16, lane);
    for (int i = 0; i < 8; ++i) {
      const float msk = (mbase + i < tcnt) ? 1.f : 0.f;
      const float vA = cA[i] + b1A, vB = cB[i] + b1B;
      sumA += msk * vA; sqA += msk * vA * vA;
      sumB += msk * vB; sqB += msk * vB * vB;
    }
    __syncthreads();
  }
  atomicAdd(&sStat[nA], sumA);       atomicAdd(&sStat[256 + nA], sqA);
  atomicAdd(&sStat[nB], sumB);       atomicAdd(&sStat[256 + nB], sqB);
  __syncthreads();
  if (t < 256) {
    const float cnt = (float)max(end - base, 1);
    const float mean = sStat[t] / cnt;
    float var = sStat[256 + t] / cnt - mean * mean;
    var = fmaxf(var, 0.f);
    mean1[g * 256 + t] = mean;
    rstd1[g * 256 + t] = rsqrtf(var + EPSI);
  }
}

// ---------------------------------------------------------------------------
// S2: per-graph stats of x2 = relu(norm1(x1)) @ W2 + b2   (64 channels)
// ---------------------------------------------------------------------------
__global__ __launch_bounds__(256) void k_stats2(
    const bf16* __restrict__ embB, const bf16* __restrict__ w1tG,
    const bf16* __restrict__ w2tG, const float* __restrict__ b1,
    const float* __restrict__ b2, const int* __restrict__ sCol,
    const int* __restrict__ sRow, const int* __restrict__ offs,
    const float* __restrict__ mean1, const float* __restrict__ rstd1,
    float* mean2, float* rstd2) {
  __shared__ __align__(32) bf16 sW1T[256 * 128];
  __shared__ __align__(32) bf16 sW2T[64 * 256];
  __shared__ __align__(32) bf16 sA[16 * 128];
  __shared__ __align__(32) bf16 sX[16 * 256];
  __shared__ float sStat[128];
  const int t = threadIdx.x, g = blockIdx.x;
  for (int i = t; i < 256 * 128 / 8; i += 256)
    ((uint4*)sW1T)[i] = ((const uint4*)w1tG)[i];
  for (int i = t; i < 64 * 256 / 8; i += 256)
    ((uint4*)sW2T)[i] = ((const uint4*)w2tG)[i];
  if (t < 128) sStat[t] = 0.f;
  __syncthreads();

  const int base = offs[g], end = offs[g + 1];
  const int wave = t >> 5, lane = t & 31;
  const int nA = wave * 32 + (lane & 15), nB = nA + 16;
  const float b1A = b1[nA], b1B = b1[nB];
  const float m1A = mean1[g * 256 + nA], r1A = rstd1[g * 256 + nA];
  const float m1B = mean1[g * 256 + nB], r1B = rstd1[g * 256 + nB];
  const int n2 = (wave & 3) * 16 + (lane & 15);
  const float b2n = b2[n2];
  const int mbase = (lane < 16) ? 0 : 8;
  float sum2 = 0.f, sq2 = 0.f;

  for (int tb = base; tb < end; tb += 16) {
    const int tcnt = min(16, end - tb);
    stage_atile(sA, embB, sCol, sRow, tb, end, t);
    __syncthreads();
    for (int j = 0; j < 2; ++j) {
      v8f c = wmma_tile(sA, sW1T, 128, wave * 32 + j * 16, lane);
      const float bv = j ? b1B : b1A, mv = j ? m1B : m1A, rv = j ? r1B : r1A;
      const int n = j ? nB : nA;
      for (int i = 0; i < 8; ++i) {
        float v = (c[i] + bv - mv) * rv;
        v = fmaxf(v, 0.f);
        sX[(mbase + i) * 256 + n] = (bf16)v;
      }
    }
    __syncthreads();
    if (wave < 4) {
      v8f c2 = wmma_tile(sX, sW2T, 256, wave * 16, lane);
      for (int i = 0; i < 8; ++i) {
        const float msk = (mbase + i < tcnt) ? 1.f : 0.f;
        const float v = c2[i] + b2n;
        sum2 += msk * v; sq2 += msk * v * v;
      }
    }
    __syncthreads();
  }
  if (wave < 4) { atomicAdd(&sStat[n2], sum2); atomicAdd(&sStat[64 + n2], sq2); }
  __syncthreads();
  if (t < 64) {
    const float cnt = (float)max(end - base, 1);
    const float mean = sStat[t] / cnt;
    float var = sStat[64 + t] / cnt - mean * mean;
    var = fmaxf(var, 0.f);
    mean2[g * 64 + t] = mean;
    rstd2[g * 64 + t] = rsqrtf(var + EPSI);
  }
}

// ---------------------------------------------------------------------------
// F: full forward, scatter to original edge order. E % 16 == 0 -> full tiles.
// ---------------------------------------------------------------------------
__global__ __launch_bounds__(256) void k_final(
    const bf16* __restrict__ embB, const bf16* __restrict__ w1tG,
    const bf16* __restrict__ w2tG, const float* __restrict__ b1,
    const float* __restrict__ b2, const float* __restrict__ W3,
    const float* __restrict__ b3, const int* __restrict__ sCol,
    const int* __restrict__ sRow, const int* __restrict__ segS,
    const int* __restrict__ perm, const float* __restrict__ mean1,
    const float* __restrict__ rstd1, const float* __restrict__ mean2,
    const float* __restrict__ rstd2, float* __restrict__ out) {
  __shared__ __align__(32) bf16 sW1T[256 * 128];
  __shared__ __align__(32) bf16 sW2T[64 * 256];
  __shared__ __align__(32) bf16 sA[16 * 128];
  __shared__ __align__(32) bf16 sX[16 * 256];
  __shared__ float sS[64 * 16];
  __shared__ int sSeg[16];
  const int t = threadIdx.x;
  for (int i = t; i < 256 * 128 / 8; i += 256)
    ((uint4*)sW1T)[i] = ((const uint4*)w1tG)[i];
  for (int i = t; i < 64 * 256 / 8; i += 256)
    ((uint4*)sW2T)[i] = ((const uint4*)w2tG)[i];
  __syncthreads();

  const int wave = t >> 5, lane = t & 31;
  const int nA = wave * 32 + (lane & 15), nB = nA + 16;
  const float b1A = b1[nA], b1B = b1[nB];
  const int n2 = (wave & 3) * 16 + (lane & 15);
  const float b2n = b2[n2];
  const float w3n = W3[n2];
  const float b3v = b3[0];
  const int mbase = (lane < 16) ? 0 : 8;
  const int nTiles = E_EDGES / 16;

  for (int tile = blockIdx.x; tile < nTiles; tile += gridDim.x) {
    const int base = tile * 16;
    if (t < 16) sSeg[t] = segS[base + t];
    stage_atile(sA, embB, sCol, sRow, base, E_EDGES, t);
    __syncthreads();
    for (int j = 0; j < 2; ++j) {
      v8f c = wmma_tile(sA, sW1T, 128, wave * 32 + j * 16, lane);
      const float bv = j ? b1B : b1A;
      const int n = j ? nB : nA;
      for (int i = 0; i < 8; ++i) {
        const int g = sSeg[mbase + i];
        float v = (c[i] + bv - mean1[g * 256 + n]) * rstd1[g * 256 + n];
        v = fmaxf(v, 0.f);
        sX[(mbase + i) * 256 + n] = (bf16)v;
      }
    }
    __syncthreads();
    if (wave < 4) {
      v8f c2 = wmma_tile(sX, sW2T, 256, wave * 16, lane);
      for (int i = 0; i < 8; ++i) {
        const int m = mbase + i;
        const int g = sSeg[m];
        float v = (c2[i] + b2n - mean2[g * 64 + n2]) * rstd2[g * 64 + n2];
        v = fmaxf(v, 0.f);
        sS[n2 * 16 + m] = v * w3n;
      }
    }
    __syncthreads();
    if (t < 16) {
      float acc = b3v;
      for (int n = 0; n < 64; ++n) acc += sS[n * 16 + t];
      out[perm[base + t]] = acc;
    }
    __syncthreads();
  }
}

// ---------------------------------------------------------------------------
extern "C" void kernel_launch(void* const* d_in, const int* in_sizes, int n_in,
                              void* d_out, int out_size, void* d_ws, size_t ws_size,
                              hipStream_t stream) {
  const float* emb   = (const float*)d_in[0];
  const float* W1    = (const float*)d_in[1];
  const float* b1    = (const float*)d_in[2];
  const float* W2    = (const float*)d_in[3];
  const float* b2    = (const float*)d_in[4];
  const float* W3    = (const float*)d_in[5];
  const float* b3    = (const float*)d_in[6];
  const int*   ei    = (const int*)d_in[7];
  const int*   batch = (const int*)d_in[8];
  float* out = (float*)d_out;

  char* p = (char*)d_ws;
  size_t off = 0;
  auto carve = [&](size_t bytes) {
    void* r = p + off;
    off += (bytes + 255) & ~(size_t)255;
    return r;
  };
  bf16* embB   = (bf16*)carve((size_t)NNODES * 64 * sizeof(bf16));
  bf16* w1t    = (bf16*)carve(256 * 128 * sizeof(bf16));
  bf16* w2t    = (bf16*)carve(64 * 256 * sizeof(bf16));
  float* mean1 = (float*)carve((size_t)NGRAPH * 256 * sizeof(float));
  float* rstd1 = (float*)carve((size_t)NGRAPH * 256 * sizeof(float));
  float* mean2 = (float*)carve((size_t)NGRAPH * 64 * sizeof(float));
  float* rstd2 = (float*)carve((size_t)NGRAPH * 64 * sizeof(float));
  int* counts  = (int*)carve(NGRAPH * sizeof(int));
  int* offs    = (int*)carve((NGRAPH + 1) * sizeof(int));
  int* cursor  = (int*)carve(NGRAPH * sizeof(int));
  int* perm    = (int*)carve((size_t)E_EDGES * sizeof(int));
  int* segS    = (int*)carve((size_t)E_EDGES * sizeof(int));
  int* sCol    = (int*)carve((size_t)E_EDGES * sizeof(int));
  int* sRow    = (int*)carve((size_t)E_EDGES * sizeof(int));

  k_prep<<<1024, 256, 0, stream>>>(emb, W1, W2, embB, w1t, w2t, counts);
  k_count<<<2048, 256, 0, stream>>>(ei, batch, counts);
  k_scan<<<1, 512, 0, stream>>>(counts, offs, cursor);
  k_scatter<<<2048, 256, 0, stream>>>(ei, batch, cursor, perm, segS, sCol, sRow);
  k_stats1<<<NGRAPH, 256, 0, stream>>>(embB, w1t, b1, sCol, sRow, offs,
                                       mean1, rstd1);
  k_stats2<<<NGRAPH, 256, 0, stream>>>(embB, w1t, w2t, b1, b2, sCol, sRow, offs,
                                       mean1, rstd1, mean2, rstd2);
  k_final<<<2048, 256, 0, stream>>>(embB, w1t, w2t, b1, b2, W3, b3, sCol, sRow,
                                    segS, perm, mean1, rstd1, mean2, rstd2, out);
}
